// SSD_76965813944448
// MI455X (gfx1250) — matrix operand order repaired and checked by reading, not compiled, not run on hardware
//
#include <hip/hip_runtime.h>
#include <hip/hip_bf16.h>
#include <cmath>

typedef __attribute__((ext_vector_type(16))) _Float16 v16h;
typedef __attribute__((ext_vector_type(8)))  _Float16 v8h;
typedef __attribute__((ext_vector_type(8)))  float    v8f;

#define TILE_M 64    // block M tile (Cout)
#define TILE_N 128   // block N tile (B*Hout*Wout)
#define LDA 40       // f16 stride of A tile row (80B; 16B-aligned sub-chunks)
#define LDB 48       // f16 stride of B tile col (96B; 16B/32B-aligned sub-chunks)

struct AnchWH { float w[6]; float h[6]; };

// ---------------- utility kernels ----------------

__global__ void cvt_f32_f16(const float* __restrict__ s, _Float16* __restrict__ d, int n) {
  int i = blockIdx.x * 256 + threadIdx.x;
  if (i < n) d[i] = (_Float16)s[i];
}

// images NCHW (fp32) -> CBHW (f16)
__global__ void img_nchw_to_cbhw(const float* __restrict__ s, _Float16* __restrict__ d,
                                 int B, int C, int H, int W) {
  int i = blockIdx.x * 256 + threadIdx.x;
  int tot = B * C * H * W;
  if (i >= tot) return;
  int x = i % W; int t = i / W;
  int y = t % H; t /= H;
  int c = t % C; int b = t / C;
  d[(((size_t)c * B + b) * H + y) * W + x] = (_Float16)s[i];
}

// ---------------- implicit-GEMM conv with WMMA (double-buffered LDS) ----------------
// X: [Cin][B][Hin][Win] f16 ; Wt: [Cout][Cin*K*K] f16 ; bias: f32
// Yh: optional f16 out [Cout][B][Hout][Wout] ; Yf: optional f32 out (same layout)
// Block: 256 threads = 8 waves; wave (wm,wn) computes a 32x32 tile of the
// 64(M) x 128(N) block tile via 4 accumulators (2 A-frags x 2 B-frags).
// K-chunk k+1 is staged into the alternate LDS buffer before the WMMAs of
// chunk k issue -> one s_barrier per K-step, global latency hidden under MMA.
__global__ __launch_bounds__(256)
void conv_wmma_kernel(const _Float16* __restrict__ X, const _Float16* __restrict__ Wt,
                      const float* __restrict__ bias,
                      _Float16* __restrict__ Yh, float* __restrict__ Yf,
                      int Cin, int Cout, int B, int Hin, int Win,
                      int Hout, int Wout, int K, int stride, int pad, int dil, int relu)
{
  __shared__ __align__(32) _Float16 sA[2][TILE_M * LDA];
  __shared__ __align__(32) _Float16 sB[2][TILE_N * LDB];

  const int tid  = threadIdx.x;
  const int wave = tid >> 5;
  const int lane = tid & 31;
  const int wm = wave >> 2;       // 0..1  (M 32-row group)
  const int wn = wave & 3;        // 0..3  (N 32-col group)
  const int HW = Hout * Wout;
  const int Ntot = B * HW;
  const int n0 = blockIdx.x * TILE_N;
  const int m0 = blockIdx.y * TILE_M;
  const int KK = K * K;
  const int Ktot = Cin * KK;

  // ---- per-thread staging assignments (fixed across the K loop) ----
  // A: row am = tid>>2 (64 rows), k-chunk akb = (tid&3)*8  (8 contiguous f16)
  const int am  = tid >> 2;
  const int akb = (tid & 3) * 8;
  const int gm_st = m0 + am;
  const _Float16* wrow = Wt + (size_t)gm_st * Ktot;

  // B: row bn = tid>>1 (128 n-values), k-chunk bkb = (tid&1)*16
  const int bn  = tid >> 1;
  const int bkb = (tid & 1) * 16;
  const int gn_st = n0 + bn;
  const bool nvalid = gn_st < Ntot;
  int b_ = 0, oy_ = 0, ox_ = 0;
  if (nvalid) {
    b_ = gn_st / HW; int rn = gn_st - b_ * HW;
    oy_ = rn / Wout; ox_ = rn - oy_ * Wout;
  }
  const int iy0 = oy_ * stride - pad;
  const int ix0 = ox_ * stride - pad;
  const size_t nbase  = (size_t)b_ * Hin * Win;
  const size_t planeB = (size_t)B * Hin * Win;   // cin stride in X

  auto stage = [&](int k0, int buf) {
    // ---- A tile: 64(m) x 32(k), one 16B vector per thread ----
    int gkb = k0 + akb;
    if (gm_st < Cout && gkb + 8 <= Ktot) {
      *(v8h*)(&sA[buf][am * LDA + akb]) = *(const v8h*)(wrow + gkb);
      __builtin_prefetch(wrow + gkb + 64, 0, 1);   // global_prefetch next K chunks
    } else {
#pragma unroll
      for (int j = 0; j < 8; ++j) {
        int gk = gkb + j;
        _Float16 v = (_Float16)0.f;
        if (gm_st < Cout && gk < Ktot) v = wrow[gk];
        sA[buf][am * LDA + akb + j] = v;
      }
    }
    // ---- B tile: 128(n) x 32(k) im2col gather; commit as 2x ds_store_b128 ----
    union { v8h v[2]; _Float16 h[16]; } bb;
#pragma unroll
    for (int j = 0; j < 16; ++j) {
      int gk = k0 + bkb + j;
      _Float16 v = (_Float16)0.f;
      if (nvalid && gk < Ktot) {
        int cin, iy, ix;
        if (K == 1) {                       // uniform branch
          cin = gk; iy = iy0; ix = ix0;
        } else if (K == 3) {                // literal divisors -> magic multiply
          cin = gk / 9; int r = gk - cin * 9;
          int ky = r / 3; int kx = r - ky * 3;
          iy = iy0 + ky * dil; ix = ix0 + kx * dil;
        } else {
          cin = gk / KK; int r = gk - cin * KK;
          int ky = r / K; int kx = r - ky * K;
          iy = iy0 + ky * dil; ix = ix0 + kx * dil;
        }
        if (iy >= 0 && iy < Hin && ix >= 0 && ix < Win)
          v = X[(size_t)cin * planeB + nbase + (size_t)iy * Win + ix];
      }
      bb.h[j] = v;
    }
    *(v8h*)(&sB[buf][bn * LDB + bkb])     = bb.v[0];
    *(v8h*)(&sB[buf][bn * LDB + bkb + 8]) = bb.v[1];
  };

  v8f acc00 = {}, acc01 = {}, acc10 = {}, acc11 = {};
  const int half = lane >> 4;
  const int l15  = lane & 15;
  const int nk = (Ktot + 31) >> 5;

  stage(0, 0);
  __syncthreads();

  for (int kc = 0; kc < nk; ++kc) {
    const int buf = kc & 1;
    if (kc + 1 < nk) stage((kc + 1) * 32, buf ^ 1);   // overlap with MMA below

    // A fragments (16x32): lane=M, lanes>=16 hold K=16..31 (K contiguous, matches B)
    const _Float16* ap0 = &sA[buf][(wm * 32 + l15) * LDA];
    const _Float16* ap1 = &sA[buf][(wm * 32 + 16 + l15) * LDA];
    v8h a0lo = *(const v8h*)(ap0 + half * 16);
    v8h a0hi = *(const v8h*)(ap0 + half * 16 + 8);
    v8h a1lo = *(const v8h*)(ap1 + half * 16);
    v8h a1hi = *(const v8h*)(ap1 + half * 16 + 8);
    v16h a0 = __builtin_shufflevector(a0lo, a0hi, 0,1,2,3,4,5,6,7,8,9,10,11,12,13,14,15);
    v16h a1 = __builtin_shufflevector(a1lo, a1hi, 0,1,2,3,4,5,6,7,8,9,10,11,12,13,14,15);
    // B fragments (32x16): lane=N, lanes>=16 hold K=16..31 (K contiguous in LDS)
    v16h b0 = *(const v16h*)(&sB[buf][(wn * 32 + l15) * LDB + half * 16]);
    v16h b1 = *(const v16h*)(&sB[buf][(wn * 32 + 16 + l15) * LDB + half * 16]);

    acc00 = __builtin_amdgcn_wmma_f32_16x16x32_f16(false, a0, false, b0, (short)0, acc00, false, false);
    acc01 = __builtin_amdgcn_wmma_f32_16x16x32_f16(false, a0, false, b1, (short)0, acc01, false, false);
    acc10 = __builtin_amdgcn_wmma_f32_16x16x32_f16(false, a1, false, b0, (short)0, acc10, false, false);
    acc11 = __builtin_amdgcn_wmma_f32_16x16x32_f16(false, a1, false, b1, (short)0, acc11, false, false);

    if (kc + 1 < nk) __syncthreads();   // next buffer staged everywhere; this buffer free
  }

  // ---- epilogue: D layout — VGPR r: lanes 0-15 M=r, lanes 16-31 M=r+8 ----
  const int gnb = n0 + wn * 32 + l15;
  const int mb  = m0 + wm * 32;

  auto sink = [&](const v8f& acc, int mbase, int gn) {
    if (gn >= Ntot) return;
#pragma unroll
    for (int r = 0; r < 8; ++r) {
      int gm = mbase + r + half * 8;
      if (gm < Cout) {
        float v = acc[r] + bias[gm];
        if (relu) v = fmaxf(v, 0.f);
        size_t o = (size_t)gm * Ntot + gn;
        if (Yh) Yh[o] = (_Float16)v;
        if (Yf) Yf[o] = v;
      }
    }
  };
  sink(acc00, mb,      gnb);
  sink(acc01, mb,      gnb + 16);
  sink(acc10, mb + 16, gnb);
  sink(acc11, mb + 16, gnb + 16);
}

// ---------------- maxpool (handles pad + ceil via OOB skip) ----------------
__global__ void maxpool_kernel(const _Float16* __restrict__ X, _Float16* __restrict__ Y,
                               int C, int B, int Hin, int Win, int Hout, int Wout,
                               int k, int s, int pad)
{
  int i = blockIdx.x * 256 + threadIdx.x;
  int tot = C * B * Hout * Wout;
  if (i >= tot) return;
  int ox = i % Wout; int t = i / Wout;
  int oy = t % Hout; t /= Hout;
  int b = t % B; int c = t / B;
  float m = -__builtin_inff();
  for (int ky = 0; ky < k; ++ky) {
    int iy = oy * s - pad + ky;
    if (iy < 0 || iy >= Hin) continue;
    for (int kx = 0; kx < k; ++kx) {
      int ix = ox * s - pad + kx;
      if (ix < 0 || ix >= Win) continue;
      float v = (float)X[(((size_t)c * B + b) * Hin + iy) * Win + ix];
      m = fmaxf(m, v);
    }
  }
  Y[i] = (_Float16)m;
}

// ---------------- conv4_3 L2 norm + learned rescale ----------------
__global__ void l2norm_kernel(const _Float16* __restrict__ X, const float* __restrict__ scale,
                              _Float16* __restrict__ Y, int C, int N)
{
  int n = blockIdx.x * 256 + threadIdx.x;
  if (n >= N) return;
  float s = 0.f;
  for (int c = 0; c < C; ++c) { float v = (float)X[(size_t)c * N + n]; s += v * v; }
  float inv = 1.f / fmaxf(sqrtf(s), 1e-12f);
  for (int c = 0; c < C; ++c)
    Y[(size_t)c * N + n] = (_Float16)((float)X[(size_t)c * N + n] * scale[c] * inv);
}

// ---------------- head cls scatter: [A*21][B][H][W] -> out[B][row][0..20] ----------------
__global__ void cls_scatter_kernel(const float* __restrict__ Yc, float* __restrict__ out,
                                   int B, int H, int W, int A, int rowBase, int rowsPerB)
{
  int i = blockIdx.x * 256 + threadIdx.x;
  int tot = B * H * W * A * 21;
  if (i >= tot) return;
  int c = i % 21; int t = i / 21;
  int a = t % A; t /= A;
  int x = t % W; t /= W;
  int y = t % H; int b = t / H;
  float v = Yc[(((size_t)(a * 21 + c) * B + b) * H + y) * W + x];
  out[((size_t)b * rowsPerB + rowBase + (y * W + x) * A + a) * 25 + c] = v;
}

// ---------------- head box decode: default boxes + regression -> out[...][21..24] ----------
__global__ void box_decode_kernel(const float* __restrict__ Yb, float* __restrict__ out,
                                  int B, int H, int W, int A, int rowBase, int rowsPerB,
                                  AnchWH wh)
{
  int i = blockIdx.x * 256 + threadIdx.x;
  int tot = B * H * W * A;
  if (i >= tot) return;
  int a = i % A; int t = i / A;
  int x = t % W; t /= W;
  int y = t % H; int b = t / H;
  size_t plane = (size_t)B * H * W;
  size_t pix = ((size_t)b * H + y) * W + x;
  float dx = Yb[(size_t)(a * 4 + 0) * plane + pix] * 0.1f;
  float dy = Yb[(size_t)(a * 4 + 1) * plane + pix] * 0.1f;
  float dw = Yb[(size_t)(a * 4 + 2) * plane + pix] * 0.2f;
  float dh = Yb[(size_t)(a * 4 + 3) * plane + pix] * 0.2f;
  float cx = (x + 0.5f) / W, cy = (y + 0.5f) / H;
  float w0 = wh.w[a], h0 = wh.h[a];
  float pcx = dx * w0 + cx, pcy = dy * h0 + cy;
  float pw = expf(dw) * w0, ph = expf(dh) * h0;
  float* o = &out[((size_t)b * rowsPerB + rowBase + (y * W + x) * A + a) * 25 + 21];
  o[0] = pcx - 0.5f * pw; o[1] = pcy - 0.5f * ph;
  o[2] = pcx + 0.5f * pw; o[3] = pcy + 0.5f * ph;
}

// ---------------- host orchestration ----------------

extern "C" void kernel_launch(void* const* d_in, const int* in_sizes, int n_in,
                              void* d_out, int out_size, void* d_ws, size_t ws_size,
                              hipStream_t stream)
{
  (void)in_sizes; (void)n_in; (void)out_size; (void)ws_size;
  const int B = 8;
  struct Spec { int cin, cout, k; };
  static const Spec SP[35] = {
    {3,64,3},{64,64,3},{64,128,3},{128,128,3},{128,256,3},{256,256,3},{256,256,3},
    {256,512,3},{512,512,3},{512,512,3},                // f0..f9
    {512,512,3},{512,512,3},{512,512,3},                // c5a..c5c
    {512,1024,3},{1024,1024,1},                         // fc6, fc7
    {1024,256,1},{256,512,3},                           // e8a, e8b
    {512,128,1},{128,256,3},                            // e9a, e9b
    {256,128,1},{128,256,3},                            // e10a, e10b
    {256,128,1},{128,256,3},                            // e11a, e11b
    {512,84,3},{1024,126,3},{512,126,3},{256,126,3},{256,84,3},{256,84,3},   // cls0..5
    {512,16,3},{1024,24,3},{512,24,3},{256,24,3},{256,16,3},{256,16,3}       // box0..5
  };

  char* base = (char*)d_ws;
  size_t off = 0;
  auto carve = [&](size_t bytes) -> void* {
    void* p = base + off;
    off += (bytes + 255) & ~(size_t)255;
    return p;
  };

  _Float16* wf[35];
  for (int i = 0; i < 35; ++i) {
    size_t n = (size_t)SP[i].cout * SP[i].cin * SP[i].k * SP[i].k;
    wf[i] = (_Float16*)carve(n * 2);
  }
  _Float16* imgH  = (_Float16*)carve((size_t)B * 3 * 300 * 300 * 2);
  _Float16* bufA  = (_Float16*)carve((size_t)64 * B * 300 * 300 * 2);
  _Float16* bufB  = (_Float16*)carve((size_t)64 * B * 300 * 300 * 2);
  _Float16* feat1 = (_Float16*)carve((size_t)512  * B * 38 * 38 * 2);
  _Float16* feat2 = (_Float16*)carve((size_t)1024 * B * 19 * 19 * 2);
  _Float16* feat3 = (_Float16*)carve((size_t)512  * B * 10 * 10 * 2);
  _Float16* feat4 = (_Float16*)carve((size_t)256  * B * 5  * 5  * 2);
  _Float16* feat5 = (_Float16*)carve((size_t)256  * B * 3  * 3  * 2);
  _Float16* feat6 = (_Float16*)carve((size_t)256  * B * 1  * 1  * 2);
  float* headBuf  = (float*)carve((size_t)1024 * 1024 * 4);

  // weight + image conversion (every call: deterministic, graph-capturable)
  for (int i = 0; i < 35; ++i) {
    int n = SP[i].cout * SP[i].cin * SP[i].k * SP[i].k;
    cvt_f32_f16<<<(n + 255) / 256, 256, 0, stream>>>((const float*)d_in[1 + 2 * i], wf[i], n);
  }
  {
    int n = B * 3 * 300 * 300;
    img_nchw_to_cbhw<<<(n + 255) / 256, 256, 0, stream>>>((const float*)d_in[0], imgH, B, 3, 300, 300);
  }
  const float* scaleW = (const float*)d_in[1 + 2 * 35];  // scale_weight [512]

  auto convX = [&](const _Float16* X, int Cin, int Hin, int Win, int si,
                   _Float16* dstH, float* dstF, int stride, int pad, int dil, int relu,
                   int* HoOut, int* WoOut) {
    int K  = SP[si].k;
    int Ho = (Hin + 2 * pad - dil * (K - 1) - 1) / stride + 1;
    int Wo = (Win + 2 * pad - dil * (K - 1) - 1) / stride + 1;
    int Ntot = B * Ho * Wo;
    dim3 g((Ntot + TILE_N - 1) / TILE_N, (SP[si].cout + TILE_M - 1) / TILE_M);
    conv_wmma_kernel<<<g, 256, 0, stream>>>(X, wf[si], (const float*)d_in[2 + 2 * si],
        dstH, dstF, Cin, SP[si].cout, B, Hin, Win, Ho, Wo, K, stride, pad, dil, relu);
    if (HoOut) *HoOut = Ho;
    if (WoOut) *WoOut = Wo;
  };

  auto pool = [&](const _Float16* X, int C, int Hin, int Win, int k, int s, int pad,
                  bool ceilm, _Float16* Y, int* HoOut) {
    int rem = (Hin + 2 * pad - k) % s;
    int extra = ceilm ? ((s - rem) % s) : 0;
    int Ho = (Hin + 2 * pad + extra - k) / s + 1;
    int tot = C * B * Ho * Ho;
    maxpool_kernel<<<(tot + 255) / 256, 256, 0, stream>>>(X, Y, C, B, Hin, Win, Ho, Ho, k, s, pad);
    *HoOut = Ho;
  };

  int Ho, Wo;
  // VGG trunk
  convX(imgH, 3,   300, 300, 0, bufA, nullptr, 1, 1, 1, 1, &Ho, &Wo);   // f0  -> 300
  convX(bufA, 64,  300, 300, 1, bufB, nullptr, 1, 1, 1, 1, &Ho, &Wo);   // f1
  pool (bufB, 64,  300, 300, 2, 2, 0, false, bufA, &Ho);                // -> 150
  convX(bufA, 64,  150, 150, 2, bufB, nullptr, 1, 1, 1, 1, &Ho, &Wo);   // f2
  convX(bufB, 128, 150, 150, 3, bufA, nullptr, 1, 1, 1, 1, &Ho, &Wo);   // f3
  pool (bufA, 128, 150, 150, 2, 2, 0, false, bufB, &Ho);                // -> 75
  convX(bufB, 128, 75, 75, 4, bufA, nullptr, 1, 1, 1, 1, &Ho, &Wo);     // f4
  convX(bufA, 256, 75, 75, 5, bufB, nullptr, 1, 1, 1, 1, &Ho, &Wo);     // f5
  convX(bufB, 256, 75, 75, 6, bufA, nullptr, 1, 1, 1, 1, &Ho, &Wo);     // f6
  pool (bufA, 256, 75, 75, 2, 2, 0, true, bufB, &Ho);                   // ceil -> 38
  convX(bufB, 256, 38, 38, 7, bufA, nullptr, 1, 1, 1, 1, &Ho, &Wo);     // f7
  convX(bufA, 512, 38, 38, 8, bufB, nullptr, 1, 1, 1, 1, &Ho, &Wo);     // f8
  convX(bufB, 512, 38, 38, 9, bufA, nullptr, 1, 1, 1, 1, &Ho, &Wo);     // f9 = conv4_3
  {
    int N = B * 38 * 38;
    l2norm_kernel<<<(N + 255) / 256, 256, 0, stream>>>(bufA, scaleW, feat1, 512, N);
  }
  pool (bufA, 512, 38, 38, 2, 2, 0, false, bufB, &Ho);                  // -> 19
  convX(bufB, 512, 19, 19, 10, bufA, nullptr, 1, 1, 1, 1, &Ho, &Wo);    // c5a
  convX(bufA, 512, 19, 19, 11, bufB, nullptr, 1, 1, 1, 1, &Ho, &Wo);    // c5b
  convX(bufB, 512, 19, 19, 12, bufA, nullptr, 1, 1, 1, 1, &Ho, &Wo);    // c5c
  pool (bufA, 512, 19, 19, 3, 1, 1, false, bufB, &Ho);                  // -> 19
  convX(bufB, 512,  19, 19, 13, bufA,  nullptr, 1, 6, 6, 1, &Ho, &Wo);  // fc6 (dil 6)
  convX(bufA, 1024, 19, 19, 14, feat2, nullptr, 1, 0, 1, 1, &Ho, &Wo);  // fc7
  convX(feat2,1024, 19, 19, 15, bufA,  nullptr, 1, 0, 1, 1, &Ho, &Wo);  // e8a
  convX(bufA, 256,  19, 19, 16, feat3, nullptr, 2, 1, 1, 1, &Ho, &Wo);  // e8b -> 10
  convX(feat3,512,  10, 10, 17, bufA,  nullptr, 1, 0, 1, 1, &Ho, &Wo);  // e9a
  convX(bufA, 128,  10, 10, 18, feat4, nullptr, 2, 1, 1, 1, &Ho, &Wo);  // e9b -> 5
  convX(feat4,256,  5,  5,  19, bufA,  nullptr, 1, 0, 1, 1, &Ho, &Wo);  // e10a
  convX(bufA, 128,  5,  5,  20, feat5, nullptr, 1, 0, 1, 1, &Ho, &Wo);  // e10b -> 3
  convX(feat5,256,  3,  3,  21, bufA,  nullptr, 1, 0, 1, 1, &Ho, &Wo);  // e11a
  convX(bufA, 128,  3,  3,  22, feat6, nullptr, 1, 0, 1, 1, &Ho, &Wo);  // e11b -> 1

  // heads
  const _Float16* F[6] = {feat1, feat2, feat3, feat4, feat5, feat6};
  const int fC[6]      = {512, 1024, 512, 256, 256, 256};
  const int fH[6]      = {38, 19, 10, 5, 3, 1};
  const int NAa[6]     = {4, 6, 6, 6, 4, 4};
  const int rowBase[6] = {0, 5776, 7942, 8542, 8692, 8728};
  const float Sc[7] = {0.1f, 0.2f, 0.375f, 0.55f, 0.725f, 0.9f, 1.0f};
  const float arsv[6][5] = {
    {1.f, 2.f, 0.5f, 0.f, 0.f}, {1.f, 2.f, 0.5f, 3.f, 1.f/3.f},
    {1.f, 2.f, 0.5f, 3.f, 1.f/3.f}, {1.f, 2.f, 0.5f, 3.f, 1.f/3.f},
    {1.f, 2.f, 0.5f, 0.f, 0.f}, {1.f, 2.f, 0.5f, 0.f, 0.f}};
  const int arsn[6] = {3, 5, 5, 5, 3, 3};

  for (int i = 0; i < 6; ++i) {
    int Hd = fH[i];
    // classification head
    convX(F[i], fC[i], Hd, Hd, 23 + i, nullptr, headBuf, 1, 1, 1, 0, &Ho, &Wo);
    int tot = B * Hd * Hd * NAa[i] * 21;
    cls_scatter_kernel<<<(tot + 255) / 256, 256, 0, stream>>>(
        headBuf, (float*)d_out, B, Hd, Hd, NAa[i], rowBase[i], 8732);
    // box head + decode with default boxes
    convX(F[i], fC[i], Hd, Hd, 29 + i, nullptr, headBuf, 1, 1, 1, 0, &Ho, &Wo);
    AnchWH wh = {};
    float sp = sqrtf(Sc[i] * Sc[i + 1]);
    wh.w[0] = sp; wh.h[0] = sp;
    for (int j = 0; j < arsn[i]; ++j) {
      float a = arsv[i][j];
      wh.w[1 + j] = Sc[i] * sqrtf(a);
      wh.h[1 + j] = Sc[i] / sqrtf(a);
    }
    tot = B * Hd * Hd * NAa[i];
    box_decode_kernel<<<(tot + 255) / 256, 256, 0, stream>>>(
        headBuf, (float*)d_out, B, Hd, Hd, NAa[i], rowBase[i], 8732, wh);
  }
}
